// ASTCG_net_30545807409353
// MI455X (gfx1250) — matrix operand order repaired and checked by reading, not compiled, not run on hardware
//
#include <hip/hip_runtime.h>
#include <hip/hip_bf16.h>

typedef __attribute__((ext_vector_type(16))) _Float16 v16h;
typedef __attribute__((ext_vector_type(4)))  _Float16 h4;
typedef __attribute__((ext_vector_type(8)))  float    v8f;

static constexpr int B_   = 16;
static constexpr int N_   = 1024;
static constexpr int CIN  = 24;
static constexpr int EMB_ = 64;
static constexpr int OUT_ = 12;
static constexpr int H_   = 2;
static constexpr int HD_  = 12;       // CIN / HEADS
static constexpr int ES_  = 12;       // EMBED_SIZE
static constexpr int HID_ = 32;       // EMB/2
static constexpr int R_   = B_ * N_;  // 16384 flattened (b,n) rows
#define ATT_SCALE 0.28867513459481287f   /* 1/sqrt(12) */

// ---------------------------------------------------------------- helpers
__device__ inline v16h hz16() {
  v16h v;
#pragma unroll
  for (int j = 0; j < 16; ++j) v[j] = (_Float16)0.f;
  return v;
}

// ---------------------------------------------------------------- utility
__global__ void zero_kernel(float* __restrict__ p, int n) {
  int i = blockIdx.x * blockDim.x + threadIdx.x;
  if (i < n) p[i] = 0.f;
}
__global__ void cvt_f16_kernel(const float* __restrict__ in, _Float16* __restrict__ out, int n) {
  int i = blockIdx.x * blockDim.x + threadIdx.x;
  if (i < n) out[i] = (_Float16)in[i];
}

// ---------------------------------------------------------------- Q/K projections (f32 math, f16 store)
__global__ void qk_kernel(const float* __restrict__ x,
                          const float* __restrict__ Wq, const float* __restrict__ bq,
                          const float* __restrict__ Wk, const float* __restrict__ bk,
                          _Float16* __restrict__ q16, _Float16* __restrict__ k16) {
  int t = blockIdx.x * blockDim.x + threadIdx.x;
  if (t >= R_ * CIN) return;
  int row = t / CIN, i = t % CIN;
  const float* xr = x + (size_t)row * CIN;
  float sq = bq[i], sk = bk[i];
#pragma unroll
  for (int c = 0; c < CIN; ++c) {
    float xv = xr[c];
    sq += xv * Wq[i * CIN + c];
    sk += xv * Wk[i * CIN + c];
  }
  q16[t] = (_Float16)sq;
  k16[t] = (_Float16)sk;
}

// ---------------------------------------------------------------- attention row max / sum-exp
// Transposed tiling: logits[key, query].  Keys ride the A operand (M dim ->
// the 8 C registers + half-wave), the 16 queries of this wave ride the B
// operand (N dim -> lanes).  Each lane owns one query; the online softmax
// update over the 8 per-tile key logits is pure per-lane VALU.  The two
// half-waves keep independent running (M,S) over their key halves, merged by
// a single shfl_xor(16) after the 64-tile loop.  Zero LDS traffic.
__global__ void attn_ms_kernel(const _Float16* __restrict__ q16,
                               const _Float16* __restrict__ k16,
                               float* __restrict__ mrow, float* __restrict__ srow) {
  const int wid = blockIdx.x;
  const int rb = wid & 63;       // N/16 = 64 query blocks
  const int bh = wid >> 6;       // b*H + h
  const int h  = bh & 1;
  const int b  = bh >> 1;
  const int lane = threadIdx.x;
  const int l = lane & 15;
  const bool hi = lane >= 16;

  // B operand (32x16): B[kk][n] = q[rb*16+n][kk].  lanes0-15 col n=l hold
  // K=0..15 (12 real); lanes16-31 cover K=16..31 -> all zero.
  v16h bq16 = hz16();
  if (!hi) {
    const _Float16* qr = q16 + ((size_t)b * N_ + rb * 16 + l) * CIN + h * HD_;
    h4 p0 = *(const h4*)qr;
    h4 p1 = *(const h4*)(qr + 4);
    h4 p2 = *(const h4*)(qr + 8);
#pragma unroll
    for (int j = 0; j < 4; ++j) { bq16[j] = p0[j]; bq16[4 + j] = p1[j]; bq16[8 + j] = p2[j]; }
  }

  float M = -3.0e38f, S = 0.f;
  for (int c0 = 0; c0 < N_; c0 += 16) {
    // A operand (16x32): row = key c0 + (lane&15).  lanes0-15: halves0..11 =
    // K0..11 (rest 0); lanes16-31: K16..31 -> all zero.
    v16h am = hz16();
    const _Float16* kr = k16 + ((size_t)b * N_ + c0 + l) * CIN + h * HD_;
    if (!hi) {
      h4 p0 = *(const h4*)kr;
      h4 p1 = *(const h4*)(kr + 4);
      h4 p2 = *(const h4*)(kr + 8);
#pragma unroll
      for (int j = 0; j < 4; ++j) { am[j] = p0[j]; am[4 + j] = p1[j]; am[8 + j] = p2[j]; }
    }
    v8f c = {};
    c = __builtin_amdgcn_wmma_f32_16x16x32_f16(false, am, false, bq16, (short)0, c, false, false);
    // C: reg v, this lane -> logit[key = c0 + v + (hi?8:0), query = l]
    float t[8];
#pragma unroll
    for (int v = 0; v < 8; ++v) t[v] = c[v] * ATT_SCALE;
    float tm = t[0];
#pragma unroll
    for (int v = 1; v < 8; ++v) tm = fmaxf(tm, t[v]);
    float nM = fmaxf(M, tm);
    float es = 0.f;
#pragma unroll
    for (int v = 0; v < 8; ++v) es += __expf(t[v] - nM);
    S = S * __expf(M - nM) + es;
    M = nM;
  }
  // merge the two half-wave key partitions (single cross-lane op)
  float Mo = __shfl_xor(M, 16, 32);
  float So = __shfl_xor(S, 16, 32);
  float nM = fmaxf(M, Mo);
  S = S * __expf(M - nM) + So * __expf(Mo - nM);
  M = nM;
  if (lane < 16) {
    int r = bh * N_ + rb * 16 + l;
    mrow[r] = M;
    srow[r] = S;
  }
}

// ---------------------------------------------------------------- sparse edge attention + weighted degrees
__global__ void edge_kernel(const int* __restrict__ ei, const float* __restrict__ ewgt,
                            const _Float16* __restrict__ q16, const _Float16* __restrict__ k16,
                            const float* __restrict__ mrow, const float* __restrict__ srow,
                            int E, float* __restrict__ deg_out, float* __restrict__ deg_in) {
  int e = blockIdx.x * blockDim.x + threadIdx.x;
  if (e >= E) return;
  int src = ei[e], dst = ei[E + e];
  float acc = 0.f;
  for (int b = 0; b < B_; ++b) {
#pragma unroll
    for (int h = 0; h < H_; ++h) {
      const _Float16* qa = q16 + ((size_t)b * N_ + src) * CIN + h * HD_;
      const _Float16* kb = k16 + ((size_t)b * N_ + dst) * CIN + h * HD_;
      float d = 0.f;
#pragma unroll
      for (int j = 0; j < HD_; ++j) d += (float)qa[j] * (float)kb[j];
      d *= ATT_SCALE;
      int idx = (b * H_ + h) * N_ + src;
      acc += __expf(d - mrow[idx]) / srow[idx];
    }
  }
  float w = 25.0f * ewgt[e] * (acc * (1.0f / (B_ * H_)));
  atomicAdd(&deg_out[src], w);
  atomicAdd(&deg_in[dst], w);
}

// ---------------------------------------------------------------- DConv message scatter (t1o / t1i)
__global__ void scatter_kernel(const int* __restrict__ ei, const float* __restrict__ x,
                               const float* __restrict__ deg_out, const float* __restrict__ deg_in,
                               int E, float* __restrict__ t1o, float* __restrict__ t1i) {
  int t = blockIdx.x * blockDim.x + threadIdx.x;
  if (t >= E * B_) return;
  int e = t >> 4, b = t & 15;
  int src = ei[e], dst = ei[E + e];
  float io = 1.0f / deg_out[src];
  float ii = 1.0f / deg_in[dst];
  const float* xs = x + ((size_t)b * N_ + src) * CIN;
  const float* xd = x + ((size_t)b * N_ + dst) * CIN;
  float* to = t1o + ((size_t)b * N_ + dst) * CIN;
  float* ti = t1i + ((size_t)b * N_ + src) * CIN;
#pragma unroll
  for (int c = 0; c < CIN; ++c) {
    atomicAdd(&to[c], xs[c] * io);
    atomicAdd(&ti[c], xd[c] * ii);
  }
}

// ---------------------------------------------------------------- weight prep for spatial WMMA
// store TRANSPOSED f16: Wt[c][k] (c = output col 0..63, k = 0..23) so the
// WMMA B-operand loads are contiguous h4 chunks per lane.
__global__ void wprep_kernel(const float* __restrict__ Wd,
                             _Float16* __restrict__ wsumT, _Float16* __restrict__ w01T,
                             _Float16* __restrict__ w11T) {
  int i = blockIdx.x * blockDim.x + threadIdx.x;   // i = k*EMB_ + c (source idx)
  const int S = CIN * EMB_;
  if (i >= S) return;
  int k = i / EMB_, c = i % EMB_;
  int o = c * CIN + k;
  wsumT[o] = (_Float16)(Wd[i] + Wd[2 * S + i]);   // Wd[0,0] + Wd[1,0]
  w01T[o]  = (_Float16)Wd[S + i];                 // Wd[0,1]
  w11T[o]  = (_Float16)Wd[3 * S + i];             // Wd[1,1]
}

// ---------------------------------------------------------------- spatial: [x|t1o|t1i] (K=24 each) x W -> 16x16 tiles
__device__ inline v8f wmma_tile24(const _Float16* __restrict__ A0,
                                  const _Float16* __restrict__ Wt, int c0,
                                  v8f acc, int l, bool hi) {
  v16h a = hz16();
  v16h bm = hz16();
  const _Float16* ar = A0 + (size_t)l * CIN;
  const _Float16* wc = Wt + (size_t)(c0 + l) * CIN;
  if (!hi) {
    h4 a0 = *(const h4*)ar, a1 = *(const h4*)(ar + 4);
    h4 a2 = *(const h4*)(ar + 16), a3 = *(const h4*)(ar + 20);
#pragma unroll
    for (int j = 0; j < 4; ++j) { a[j] = a0[j]; a[4 + j] = a1[j]; a[8 + j] = a2[j]; a[12 + j] = a3[j]; }
    h4 b0 = *(const h4*)wc, b1 = *(const h4*)(wc + 4);
    h4 b2 = *(const h4*)(wc + 16), b3 = *(const h4*)(wc + 20);
#pragma unroll
    for (int j = 0; j < 4; ++j) { bm[j] = b0[j]; bm[4 + j] = b1[j]; bm[8 + j] = b2[j]; bm[12 + j] = b3[j]; }
  } else {
    h4 a0 = *(const h4*)(ar + 8), a1 = *(const h4*)(ar + 12);
#pragma unroll
    for (int j = 0; j < 4; ++j) { a[j] = a0[j]; a[4 + j] = a1[j]; }
    h4 b0 = *(const h4*)(wc + 8), b1 = *(const h4*)(wc + 12);
#pragma unroll
    for (int j = 0; j < 4; ++j) { bm[j] = b0[j]; bm[4 + j] = b1[j]; }
  }
  return __builtin_amdgcn_wmma_f32_16x16x32_f16(false, a, false, bm, (short)0, acc, false, false);
}

__global__ void spatial_kernel(const _Float16* __restrict__ x16,
                               const _Float16* __restrict__ t1o16,
                               const _Float16* __restrict__ t1i16,
                               const _Float16* __restrict__ wsumT,
                               const _Float16* __restrict__ w01T,
                               const _Float16* __restrict__ w11T,
                               const float* __restrict__ bd,
                               float* __restrict__ spatial) {
  int wid = blockIdx.x;
  int cb = wid & 3, rb = wid >> 2;          // 4 col tiles, R_/16 row tiles
  int lane = threadIdx.x, l = lane & 15;
  bool hi = lane >= 16;
  int c0 = cb * 16;
  size_t abase = (size_t)rb * 16 * CIN;
  v8f acc = {};
  acc = wmma_tile24(x16 + abase,   wsumT, c0, acc, l, hi);
  acc = wmma_tile24(t1o16 + abase, w01T,  c0, acc, l, hi);
  acc = wmma_tile24(t1i16 + abase, w11T,  c0, acc, l, hi);
  int col = c0 + l;
  float bias = bd[col];
#pragma unroll
  for (int v = 0; v < 8; ++v) {
    int m = v + (hi ? 8 : 0);
    spatial[((size_t)rb * 16 + m) * EMB_ + col] = fmaxf(acc[v] + bias, 0.f);
  }
}

// ---------------------------------------------------------------- TCN (B,C,N), causal k=2
__global__ void tcn0_conv1(const float* __restrict__ x, const float* __restrict__ w,
                           const float* __restrict__ bias, float* __restrict__ o1) {
  int n = blockIdx.x * blockDim.x + threadIdx.x;
  int co = blockIdx.y, b = blockIdx.z;
  float acc = bias[co];
  const float* wr = w + co * CIN * 2;
#pragma unroll
  for (int ci = 0; ci < CIN; ++ci) {
    float cur  = x[((size_t)b * N_ + n) * CIN + ci];
    float past = (n >= 1) ? x[((size_t)b * N_ + n - 1) * CIN + ci] : 0.f;
    acc += wr[ci * 2] * past + wr[ci * 2 + 1] * cur;
  }
  o1[((size_t)b * EMB_ + co) * N_ + n] = fmaxf(acc, 0.f);
}

__global__ void tcn0_conv2_down(const float* __restrict__ o1, const float* __restrict__ x,
                                const float* __restrict__ w2, const float* __restrict__ b2,
                                const float* __restrict__ wd, const float* __restrict__ bdn,
                                float* __restrict__ h0) {
  int n = blockIdx.x * blockDim.x + threadIdx.x;
  int co = blockIdx.y, b = blockIdx.z;
  float acc = b2[co];
  const float* wr = w2 + co * EMB_ * 2;
  for (int ci = 0; ci < EMB_; ++ci) {
    const float* ir = o1 + ((size_t)b * EMB_ + ci) * N_;
    float cur = ir[n], past = (n >= 1) ? ir[n - 1] : 0.f;
    acc += wr[ci * 2] * past + wr[ci * 2 + 1] * cur;
  }
  acc = fmaxf(acc, 0.f);
  float dn = bdn[co];
  const float* wdr = wd + co * CIN;
#pragma unroll
  for (int ci = 0; ci < CIN; ++ci)
    dn += wdr[ci] * x[((size_t)b * N_ + n) * CIN + ci];
  h0[((size_t)b * EMB_ + co) * N_ + n] = fmaxf(acc + dn, 0.f);
}

__global__ void tcn1_conv1(const float* __restrict__ h0, const float* __restrict__ w,
                           const float* __restrict__ bias, float* __restrict__ o3) {
  int n = blockIdx.x * blockDim.x + threadIdx.x;
  int co = blockIdx.y, b = blockIdx.z;
  float acc = bias[co];
  const float* wr = w + co * EMB_ * 2;
  for (int ci = 0; ci < EMB_; ++ci) {
    const float* ir = h0 + ((size_t)b * EMB_ + ci) * N_;
    float cur = ir[n], past = (n >= 2) ? ir[n - 2] : 0.f;   // dilation 2
    acc += wr[ci * 2] * past + wr[ci * 2 + 1] * cur;
  }
  o3[((size_t)b * EMB_ + co) * N_ + n] = fmaxf(acc, 0.f);
}

__global__ void tcn1_conv2_res(const float* __restrict__ o3, const float* __restrict__ h0,
                               const float* __restrict__ w, const float* __restrict__ bias,
                               float* __restrict__ h1) {
  int n = blockIdx.x * blockDim.x + threadIdx.x;
  int co = blockIdx.y, b = blockIdx.z;
  float acc = bias[co];
  const float* wr = w + co * EMB_ * 2;
  for (int ci = 0; ci < EMB_; ++ci) {
    const float* ir = o3 + ((size_t)b * EMB_ + ci) * N_;
    float cur = ir[n], past = (n >= 2) ? ir[n - 2] : 0.f;   // dilation 2
    acc += wr[ci * 2] * past + wr[ci * 2 + 1] * cur;
  }
  size_t idx = ((size_t)b * EMB_ + co) * N_ + n;
  h1[idx] = fmaxf(fmaxf(acc, 0.f) + h0[idx], 0.f);
}

// ---------------------------------------------------------------- fuse: emb = [temporal|spatial]@We.T+be ; xp = emb@Wih.T+bih
__global__ void fuse_kernel(const float* __restrict__ h1, const float* __restrict__ spatial,
                            const float* __restrict__ We, const float* __restrict__ be,
                            const float* __restrict__ Wih, const float* __restrict__ bih,
                            float* __restrict__ xp) {
  int t = blockIdx.x * blockDim.x + threadIdx.x;
  if (t >= R_) return;
  int b = t >> 10, n = t & 1023;
  float emb[ES_];
#pragma unroll
  for (int j = 0; j < ES_; ++j) {
    float a = be[j];
    const float* wr = We + j * (2 * EMB_);
    for (int i = 0; i < EMB_; ++i) a += wr[i] * h1[((size_t)b * EMB_ + i) * N_ + n];
    const float* sp = spatial + (size_t)t * EMB_;
    for (int i = 0; i < EMB_; ++i) a += wr[EMB_ + i] * sp[i];
    emb[j] = a;
  }
  float* xpr = xp + (size_t)t * (3 * HID_);
  for (int j = 0; j < 3 * HID_; ++j) {
    float a = bih[j];
    const float* wr = Wih + j * ES_;
#pragma unroll
    for (int i = 0; i < ES_; ++i) a += wr[i] * emb[i];
    xpr[j] = a;
  }
}

// ---------------------------------------------------------------- GRU scan over N (sequential) + fused decoder
__global__ void gru_kernel(const float* __restrict__ xp,
                           const float* __restrict__ Whh, const float* __restrict__ bhh,
                           const float* __restrict__ Wdec, const float* __restrict__ bdec,
                           float* __restrict__ out) {
  __shared__ float h[B_ * HID_];           // 512
  __shared__ float gh[B_ * 3 * HID_];      // 1536
  __shared__ float sWhh[3 * HID_ * HID_];  // 3072
  __shared__ float sbhh[3 * HID_];
  __shared__ float sWdec[OUT_ * HID_];
  __shared__ float sbdec[OUT_];
  int tid = threadIdx.x;
  for (int i = tid; i < 3 * HID_ * HID_; i += 256) sWhh[i] = Whh[i];
  for (int i = tid; i < 3 * HID_; i += 256) sbhh[i] = bhh[i];
  for (int i = tid; i < OUT_ * HID_; i += 256) sWdec[i] = Wdec[i];
  if (tid < OUT_) sbdec[tid] = bdec[tid];
  for (int i = tid; i < B_ * HID_; i += 256) h[i] = 0.f;
  __syncthreads();

  for (int n = 0; n < N_; ++n) {
    // phase1: gh = h @ Whh.T + bhh
    for (int i = tid; i < B_ * 3 * HID_; i += 256) {
      int b = i / (3 * HID_), j = i % (3 * HID_);
      float a = sbhh[j];
      const float* wr = sWhh + j * HID_;
      const float* hb = h + b * HID_;
#pragma unroll
      for (int k = 0; k < HID_; ++k) a += wr[k] * hb[k];
      gh[i] = a;
    }
    __syncthreads();
    // phase2: gates + hnew
    for (int i = tid; i < B_ * HID_; i += 256) {
      int b = i / HID_, hid = i % HID_;
      const float* xr = xp + ((size_t)b * N_ + n) * (3 * HID_);
      float xr_ = xr[hid], xz_ = xr[HID_ + hid], xn_ = xr[2 * HID_ + hid];
      float hr = gh[b * 3 * HID_ + hid];
      float hz = gh[b * 3 * HID_ + HID_ + hid];
      float hn = gh[b * 3 * HID_ + 2 * HID_ + hid];
      float r = 1.f / (1.f + __expf(-(xr_ + hr)));
      float z = 1.f / (1.f + __expf(-(xz_ + hz)));
      float nn = tanhf(xn_ + r * hn);
      float hp = h[i];
      h[i] = (1.f - z) * nn + z * hp;
    }
    __syncthreads();
    // phase3: out[b,n,:] = relu(h) @ Wdec.T + bdec   (reads h only)
    for (int i = tid; i < B_ * OUT_; i += 256) {
      int b = i / OUT_, o = i % OUT_;
      float a = sbdec[o];
      const float* wr = sWdec + o * HID_;
      const float* hb = h + b * HID_;
#pragma unroll
      for (int k = 0; k < HID_; ++k) a += wr[k] * fmaxf(hb[k], 0.f);
      out[((size_t)b * N_ + n) * OUT_ + o] = a;
    }
    __syncthreads();
  }
}

// ================================================================ launch
extern "C" void kernel_launch(void* const* d_in, const int* in_sizes, int n_in,
                              void* d_out, int out_size, void* d_ws, size_t ws_size,
                              hipStream_t stream) {
  const float* x    = (const float*)d_in[0];
  const int*   ei   = (const int*)d_in[2];
  const float* ewgt = (const float*)d_in[3];
  const int    E    = in_sizes[3];
  const float* Wq = (const float*)d_in[4],  *bq = (const float*)d_in[5];
  const float* Wk = (const float*)d_in[6],  *bk = (const float*)d_in[7];
  const float* Wd = (const float*)d_in[8],  *bd = (const float*)d_in[9];
  const float* c1w0 = (const float*)d_in[10], *c1b0 = (const float*)d_in[11];
  const float* c2w0 = (const float*)d_in[12], *c2b0 = (const float*)d_in[13];
  const float* dw0  = (const float*)d_in[14], *db0  = (const float*)d_in[15];
  const float* c1w1 = (const float*)d_in[16], *c1b1 = (const float*)d_in[17];
  const float* c2w1 = (const float*)d_in[18], *c2b1 = (const float*)d_in[19];
  const float* We   = (const float*)d_in[20], *be   = (const float*)d_in[21];
  const float* Wih  = (const float*)d_in[22], *Whh  = (const float*)d_in[23];
  const float* bih  = (const float*)d_in[24], *bhh  = (const float*)d_in[25];
  const float* Wdec = (const float*)d_in[26], *bdec = (const float*)d_in[27];
  float* out = (float*)d_out;
  (void)n_in; (void)out_size; (void)ws_size;

  char* p = (char*)d_ws;
  auto alloc = [&](size_t bytes) -> char* {
    char* r = p;
    p += (bytes + 255) & ~(size_t)255;
    return r;
  };
  _Float16* q16   = (_Float16*)alloc((size_t)R_ * CIN * 2);
  _Float16* k16   = (_Float16*)alloc((size_t)R_ * CIN * 2);
  float* mrow     = (float*)alloc((size_t)B_ * H_ * N_ * 4);
  float* srow     = (float*)alloc((size_t)B_ * H_ * N_ * 4);
  // contiguous zero region: deg_out | deg_in | t1o | t1i
  const int nzero = 2 * N_ + 2 * R_ * CIN;
  float* zreg     = (float*)alloc((size_t)nzero * 4);
  float* deg_out  = zreg;
  float* deg_in   = zreg + N_;
  float* t1o      = zreg + 2 * N_;
  float* t1i      = t1o + (size_t)R_ * CIN;
  _Float16* x16   = (_Float16*)alloc((size_t)R_ * CIN * 2);
  _Float16* t1o16 = (_Float16*)alloc((size_t)R_ * CIN * 2);
  _Float16* t1i16 = (_Float16*)alloc((size_t)R_ * CIN * 2);
  _Float16* wsumT = (_Float16*)alloc((size_t)CIN * EMB_ * 2);
  _Float16* w01T  = (_Float16*)alloc((size_t)CIN * EMB_ * 2);
  _Float16* w11T  = (_Float16*)alloc((size_t)CIN * EMB_ * 2);
  float* spat     = (float*)alloc((size_t)R_ * EMB_ * 4);
  float* o1       = (float*)alloc((size_t)B_ * EMB_ * N_ * 4);
  float* h0       = (float*)alloc((size_t)B_ * EMB_ * N_ * 4);
  float* o3       = (float*)alloc((size_t)B_ * EMB_ * N_ * 4);
  float* h1       = (float*)alloc((size_t)B_ * EMB_ * N_ * 4);
  float* xp       = (float*)alloc((size_t)R_ * 3 * HID_ * 4);

  // 1. Q/K projections -> f16
  qk_kernel<<<(R_ * CIN + 255) / 256, 256, 0, stream>>>(x, Wq, bq, Wk, bk, q16, k16);
  // 2. x -> f16 (for spatial WMMA)
  cvt_f16_kernel<<<(R_ * CIN + 255) / 256, 256, 0, stream>>>(x, x16, R_ * CIN);
  // 3. attention per-row max / sum-exp via WMMA + register-space online softmax
  attn_ms_kernel<<<B_ * H_ * (N_ / 16), 32, 0, stream>>>(q16, k16, mrow, srow);
  // 4. zero accumulators (ws is not re-poisoned between replays)
  zero_kernel<<<(nzero + 255) / 256, 256, 0, stream>>>(zreg, nzero);
  // 5. sparse edge attention + weighted degrees
  edge_kernel<<<(E + 255) / 256, 256, 0, stream>>>(ei, ewgt, q16, k16, mrow, srow, E, deg_out, deg_in);
  // 6. DConv message scatter
  scatter_kernel<<<(E * B_ + 255) / 256, 256, 0, stream>>>(ei, x, deg_out, deg_in, E, t1o, t1i);
  // 7. weight prep (transposed f16) + f16 conversions
  wprep_kernel<<<(CIN * EMB_ + 255) / 256, 256, 0, stream>>>(Wd, wsumT, w01T, w11T);
  cvt_f16_kernel<<<(R_ * CIN + 255) / 256, 256, 0, stream>>>(t1o, t1o16, R_ * CIN);
  cvt_f16_kernel<<<(R_ * CIN + 255) / 256, 256, 0, stream>>>(t1i, t1i16, R_ * CIN);
  // 8. spatial = relu([x|t1o|t1i] @ W + bd) via WMMA
  spatial_kernel<<<(R_ / 16) * (EMB_ / 16), 32, 0, stream>>>(x16, t1o16, t1i16, wsumT, w01T, w11T, bd, spat);
  // 9. TCN
  dim3 cg(N_ / 256, EMB_, B_);
  tcn0_conv1<<<cg, 256, 0, stream>>>(x, c1w0, c1b0, o1);
  tcn0_conv2_down<<<cg, 256, 0, stream>>>(o1, x, c2w0, c2b0, dw0, db0, h0);
  tcn1_conv1<<<cg, 256, 0, stream>>>(h0, c1w1, c1b1, o3);
  tcn1_conv2_res<<<cg, 256, 0, stream>>>(o3, h0, c2w1, c2b1, h1);
  // 10. fuse + input projections
  fuse_kernel<<<(R_ + 255) / 256, 256, 0, stream>>>(h1, spat, We, be, Wih, bih, xp);
  // 11. GRU scan + decoder (single block, LDS-resident state)
  gru_kernel<<<1, 256, 0, stream>>>(xp, Whh, bhh, Wdec, bdec, out);
}
